// MultiHeadAttention_90855738180015
// MI455X (gfx1250) — matrix operand order repaired and checked by reading, not compiled, not run on hardware
//
#include <hip/hip_runtime.h>
#include <hip/hip_bf16.h>

// ---------------------------------------------------------------------------
// MHA forward for MI455X (gfx1250), bf16 WMMA path, 16x64 C-strip per wave.
// B=2, T=2048, D=1024, H=16, DK=64. Outputs: [output f32 BTD][attn f32 BHTT].
// Inputs/weights pre-converted to bf16 once (streaming kernel) so all GEMM
// hot loops are pure b128 loads + v_wmma. attn (512MB stream) is NT-policy.
// ---------------------------------------------------------------------------

typedef __attribute__((ext_vector_type(16))) __bf16 v16bf;
typedef __attribute__((ext_vector_type(8)))  __bf16 v8bf;
typedef __attribute__((ext_vector_type(4)))  __bf16 v4bf;
typedef __attribute__((ext_vector_type(2)))  __bf16 v2bf;
typedef __attribute__((ext_vector_type(8)))  float  v8f;
typedef __attribute__((ext_vector_type(4)))  float  v4f;

constexpr int    BB   = 2;
constexpr int    TT   = 2048;
constexpr int    DD   = 1024;
constexpr int    HH   = 16;
constexpr int    DK   = 64;
constexpr float  NEGV = -999999999.0f;
constexpr float  KEEP = 0.9f;          // 1 - dropout (eval-mode scaling)
constexpr float  SCL  = 0.125f;        // 1/sqrt(DK)

// ---- f32 -> bf16 (round to nearest even) ----------------------------------
static __device__ inline __bf16 f2bf(float f) {
    unsigned u = __builtin_bit_cast(unsigned, f);
    u += 0x7FFFu + ((u >> 16) & 1u);
    return __builtin_bit_cast(__bf16, (unsigned short)(u >> 16));
}

#if __has_builtin(__builtin_amdgcn_cvt_pk_bf16_f32)
static __device__ inline v2bf pk2(float a, float b) {
    return __builtin_amdgcn_cvt_pk_bf16_f32(a, b);
}
#else
static __device__ inline v2bf pk2(float a, float b) {
    v2bf r; r[0] = f2bf(a); r[1] = f2bf(b); return r;
}
#endif

static __device__ inline v8bf pack8(v4f x, v4f y) {
    v2bf p0 = pk2(x[0], x[1]), p1 = pk2(x[2], x[3]);
    v2bf p2 = pk2(y[0], y[1]), p3 = pk2(y[2], y[3]);
    v4bf lo = __builtin_shufflevector(p0, p1, 0, 1, 2, 3);
    v4bf hi = __builtin_shufflevector(p2, p3, 0, 1, 2, 3);
    return __builtin_shufflevector(lo, hi, 0, 1, 2, 3, 4, 5, 6, 7);
}
static __device__ inline v8bf cvt8_nt(const float* p) {
    v4f x = __builtin_nontemporal_load((const v4f*)p);
    v4f y = __builtin_nontemporal_load((const v4f*)(p + 4));
    return pack8(x, y);
}

static __device__ inline v16bf cat8(v8bf lo, v8bf hi) {
    return __builtin_shufflevector(lo, hi, 0, 1, 2, 3, 4, 5, 6, 7,
                                           8, 9, 10, 11, 12, 13, 14, 15);
}

// ---- WMMA fragment loaders (ISA 7.12.2 layouts, wave32) -------------------
// A 16x32: lane row = lane&15, half = lane>>4. Element e at
// K = 16*(e/8) + 8*half + (e%8). Caller pre-offsets base by half*8:
// the two 8-element runs sit at +0 and +16.
static __device__ inline v16bf load_a_bf16(const __bf16* base) {
    return cat8(*(const v8bf*)base, *(const v8bf*)(base + 16));
}
static __device__ inline v16bf load_a_f32_nt(const float* base) {
    return cat8(cvt8_nt(base), cvt8_nt(base + 16));
}
// B 32x16: lane col = lane&15; element e holds K = 16*(e/8) + 8*half + (e%8).
// Caller pre-offsets base to (K = 8*half, N = lane&15); runs at +0 and +16.
static __device__ inline v16bf load_b_bf16(const __bf16* base) {
    return cat8(*(const v8bf*)base, *(const v8bf*)(base + 16));
}

static __device__ inline v8f wmma_bf16(v16bf a, v16bf b, v8f c) {
    return __builtin_amdgcn_wmma_f32_16x16x32_bf16(false, a, false, b,
                                                   (short)0, c, false, false);
}

// ---------------------------------------------------------------------------
// 0) Streaming f32 -> bf16 conversion (run once per input/weight buffer).
// ---------------------------------------------------------------------------
__global__ void __launch_bounds__(256)
cvt_kernel(const float* __restrict__ src, __bf16* __restrict__ dst, int n8)
{
    const int i = blockIdx.x * 256 + threadIdx.x;
    if (i >= n8) return;
    const float* p = src + (size_t)i * 8;
    v8bf r = pack8(*(const v4f*)p, *(const v4f*)(p + 4));
    *(v8bf*)(dst + (size_t)i * 8) = r;
}

// ---------------------------------------------------------------------------
// 1) Projection: dst = X @ W^T (all bf16).  Wave computes 16x64 strip.
//    VT=false: dst[b,h,t,dk]   VT=true: dst[b,h,dk,t]  (transposed for AV B)
// ---------------------------------------------------------------------------
template <bool VT>
__global__ void __launch_bounds__(128)
proj_kernel(const __bf16* __restrict__ X, const __bf16* __restrict__ W,
            __bf16* __restrict__ dst)
{
    const int lane = threadIdx.x & 31;
    const int wid  = threadIdx.x >> 5;
    const int n0   = (blockIdx.x * 4 + wid) * 64;  // multiple of 64
    const int m0   = blockIdx.y * 16;
    const int row  = lane & 15;
    const int half = lane >> 4;

    const __bf16* ap = X + (size_t)(m0 + row) * DD + half * 8;
    const __bf16* bp = W + (size_t)(n0 + row) * DD + half * 8;

    v8f acc[4] = {};
    for (int kb = 0; kb < DD; kb += 32) {
        v16bf a  = load_a_bf16(ap);
        v16bf b0 = load_b_bf16(bp);
        v16bf b1 = load_b_bf16(bp + 16 * DD);
        v16bf b2 = load_b_bf16(bp + 32 * DD);
        v16bf b3 = load_b_bf16(bp + 48 * DD);
        acc[0] = wmma_bf16(a, b0, acc[0]);
        acc[1] = wmma_bf16(a, b1, acc[1]);
        acc[2] = wmma_bf16(a, b2, acc[2]);
        acc[3] = wmma_bf16(a, b3, acc[3]);
        ap += 32; bp += 32;
    }

    // m-strip never crosses a T boundary (m0 mult of 16, TT mult of 16)
    const int bb = m0 / TT;            // batch (wave-constant)
    const int t0 = m0 - bb * TT + 8 * half;
    const int h  = n0 / DK;            // head  (wave-constant; n0 mult of 64)
    if (VT) {
        // dst[((bb*HH+h)*DK + dk)*TT + t],  dk = nt*16 + row
        __bf16* op = dst + (((size_t)bb * HH + h) * DK + row) * TT + t0;
#pragma unroll
        for (int nt = 0; nt < 4; ++nt)
#pragma unroll
            for (int r = 0; r < 8; ++r)
                op[(size_t)nt * 16 * TT + r] = f2bf(acc[nt][r]);
    } else {
        // dst[((bb*HH+h)*TT + t)*DK + dk]
        __bf16* op = dst + (((size_t)bb * HH + h) * TT + t0) * DK + row;
#pragma unroll
        for (int nt = 0; nt < 4; ++nt)
#pragma unroll
            for (int r = 0; r < 8; ++r)
                op[r * DK + nt * 16] = f2bf(acc[nt][r]);
    }
}

// ---------------------------------------------------------------------------
// 2) Scores: attn[bh,q,k] = (q . k) * 0.125, masked -> NEG. 16x64 strip/wave.
// ---------------------------------------------------------------------------
__global__ void __launch_bounds__(128)
scores_kernel(const __bf16* __restrict__ qb, const __bf16* __restrict__ kb,
              const unsigned char* __restrict__ mask, float* __restrict__ attn)
{
    const int lane  = threadIdx.x & 31;
    const int wid   = threadIdx.x >> 5;
    const int k0    = (blockIdx.x * 4 + wid) * 64;
    const int qtile = blockIdx.y;
    const int bh    = blockIdx.z;
    const int row   = lane & 15;
    const int half  = lane >> 4;

    const __bf16* ap = qb + (size_t)bh * TT * DK
                          + (size_t)(qtile * 16 + row) * DK + half * 8;
    const __bf16* bp = kb + (size_t)bh * TT * DK
                          + (size_t)(k0 + row) * DK + half * 8;

    v8f acc[4] = {};
#pragma unroll
    for (int ks = 0; ks < DK; ks += 32) {
        v16bf a  = load_a_bf16(ap + ks);
        v16bf b0 = load_b_bf16(bp + ks);
        v16bf b1 = load_b_bf16(bp + ks + 16 * DK);
        v16bf b2 = load_b_bf16(bp + ks + 32 * DK);
        v16bf b3 = load_b_bf16(bp + ks + 48 * DK);
        acc[0] = wmma_bf16(a, b0, acc[0]);
        acc[1] = wmma_bf16(a, b1, acc[1]);
        acc[2] = wmma_bf16(a, b2, acc[2]);
        acc[3] = wmma_bf16(a, b3, acc[3]);
    }

    const int qr0 = qtile * 16 + 8 * half;
    float* op = attn + ((size_t)bh * TT + qr0) * TT + k0 + row;
    const unsigned char* mp = mask + (size_t)qr0 * TT + k0 + row;
#pragma unroll
    for (int nt = 0; nt < 4; ++nt)
#pragma unroll
        for (int r = 0; r < 8; ++r) {
            float sc = acc[nt][r] * SCL;
            sc = mp[(size_t)r * TT + nt * 16] ? NEGV : sc;
            __builtin_nontemporal_store(sc, op + (size_t)r * TT + nt * 16);
        }
}

// ---------------------------------------------------------------------------
// 3) Row softmax over attn (in place). One block per (bh, row).
// ---------------------------------------------------------------------------
__global__ void __launch_bounds__(256)
softmax_kernel(float* __restrict__ attn)
{
    const int row = blockIdx.x;
    const int bh  = blockIdx.y;
    const int tid = threadIdx.x;
    float* p = attn + ((size_t)bh * TT + row) * TT;

    float vals[TT / 256];
    float m = -3.0e38f;
#pragma unroll
    for (int i = 0; i < TT / 256; ++i) {
        vals[i] = __builtin_nontemporal_load(&p[tid + i * 256]);
        m = fmaxf(m, vals[i]);
    }
    __shared__ float red[256];
    red[tid] = m; __syncthreads();
    for (int s = 128; s > 0; s >>= 1) {
        if (tid < s) red[tid] = fmaxf(red[tid], red[tid + s]);
        __syncthreads();
    }
    m = red[0]; __syncthreads();

    float sum = 0.f;
#pragma unroll
    for (int i = 0; i < TT / 256; ++i) {
        vals[i] = __expf(vals[i] - m);
        sum += vals[i];
    }
    red[tid] = sum; __syncthreads();
    for (int s = 128; s > 0; s >>= 1) {
        if (tid < s) red[tid] += red[tid + s];
        __syncthreads();
    }
    const float inv = 1.0f / red[0];
#pragma unroll
    for (int i = 0; i < TT / 256; ++i)
        __builtin_nontemporal_store(vals[i] * inv, &p[tid + i * 256]);
}

// ---------------------------------------------------------------------------
// 4) AV: heads[b,t,h*DK+dk] = sum_k attn[bh,t,k] * v[bh,k,dk]
//    v stored transposed [bh, dk, t] so B fragments are contiguous loads.
//    Wave computes 16 q-rows x all 64 dk (attn row read exactly once, NT).
// ---------------------------------------------------------------------------
__global__ void __launch_bounds__(128)
av_kernel(const float* __restrict__ attn, const __bf16* __restrict__ vt,
          __bf16* __restrict__ heads)
{
    const int lane  = threadIdx.x & 31;
    const int wid   = threadIdx.x >> 5;
    const int qtile = blockIdx.y * 4 + wid;   // 0..127
    const int bh    = blockIdx.z;             // 0..31
    const int row   = lane & 15;
    const int half  = lane >> 4;

    const float*  ap = attn + ((size_t)bh * TT + qtile * 16 + row) * TT + half * 8;
    const __bf16* bp = vt + (size_t)bh * DK * TT + (size_t)row * TT + half * 8;

    v8f acc[4] = {};
    for (int kb = 0; kb < TT; kb += 32) {
        __builtin_prefetch(ap + 256, 0, 1);
        v16bf a  = load_a_f32_nt(ap);
        v16bf b0 = load_b_bf16(bp);
        v16bf b1 = load_b_bf16(bp + 16 * TT);
        v16bf b2 = load_b_bf16(bp + 32 * TT);
        v16bf b3 = load_b_bf16(bp + 48 * TT);
        acc[0] = wmma_bf16(a, b0, acc[0]);
        acc[1] = wmma_bf16(a, b1, acc[1]);
        acc[2] = wmma_bf16(a, b2, acc[2]);
        acc[3] = wmma_bf16(a, b3, acc[3]);
        ap += 32; bp += 32;
    }

    const int b_ = bh / HH, h = bh % HH;
    const int qr0 = qtile * 16 + 8 * half;
    __bf16* op = heads + ((size_t)b_ * TT + qr0) * DD + h * DK + row;
#pragma unroll
    for (int nt = 0; nt < 4; ++nt)
#pragma unroll
        for (int r = 0; r < 8; ++r)
            op[(size_t)r * DD + nt * 16] = f2bf(acc[nt][r]);
}

// ---------------------------------------------------------------------------
// 5) Output projection: out = (heads @ W_O^T) * KEEP.  16x64 strip per wave.
// ---------------------------------------------------------------------------
__global__ void __launch_bounds__(128)
oproj_kernel(const __bf16* __restrict__ heads, const __bf16* __restrict__ WO,
             float* __restrict__ out)
{
    const int lane = threadIdx.x & 31;
    const int wid  = threadIdx.x >> 5;
    const int n0   = (blockIdx.x * 4 + wid) * 64;
    const int m0   = blockIdx.y * 16;
    const int row  = lane & 15;
    const int half = lane >> 4;

    const __bf16* ap = heads + (size_t)(m0 + row) * DD + half * 8;
    const __bf16* bp = WO + (size_t)(n0 + row) * DD + half * 8;

    v8f acc[4] = {};
    for (int kb = 0; kb < DD; kb += 32) {
        v16bf a  = load_a_bf16(ap);
        v16bf b0 = load_b_bf16(bp);
        v16bf b1 = load_b_bf16(bp + 16 * DD);
        v16bf b2 = load_b_bf16(bp + 32 * DD);
        v16bf b3 = load_b_bf16(bp + 48 * DD);
        acc[0] = wmma_bf16(a, b0, acc[0]);
        acc[1] = wmma_bf16(a, b1, acc[1]);
        acc[2] = wmma_bf16(a, b2, acc[2]);
        acc[3] = wmma_bf16(a, b3, acc[3]);
        ap += 32; bp += 32;
    }

    float* op = out + (size_t)(m0 + 8 * half) * DD + n0 + row;
#pragma unroll
    for (int nt = 0; nt < 4; ++nt)
#pragma unroll
        for (int r = 0; r < 8; ++r)
            op[(size_t)r * DD + nt * 16] = acc[nt][r] * KEEP;
}

// ---------------------------------------------------------------------------
extern "C" void kernel_launch(void* const* d_in, const int* in_sizes, int n_in,
                              void* d_out, int out_size, void* d_ws, size_t ws_size,
                              hipStream_t stream)
{
    (void)in_sizes; (void)n_in; (void)out_size; (void)ws_size;

    const float* Q  = (const float*)d_in[0];
    const float* K  = (const float*)d_in[1];
    const float* V  = (const float*)d_in[2];
    const unsigned char* mask = (const unsigned char*)d_in[3];
    const float* WQ = (const float*)d_in[4];
    const float* WK = (const float*)d_in[5];
    const float* WV = (const float*)d_in[6];
    const float* WO = (const float*)d_in[7];

    float* out  = (float*)d_out;
    float* attn = out + (size_t)BB * TT * DD;   // output first, attn second

    char* ws = (char*)d_ws;
    const size_t MB = 1ull << 20;
    __bf16* qb    = (__bf16*)(ws);              //  8 MB [b,h,t,dk]
    __bf16* kb    = (__bf16*)(ws +  8 * MB);    //  8 MB [b,h,t,dk]
    __bf16* vt    = (__bf16*)(ws + 16 * MB);    //  8 MB [b,h,dk,t]
    __bf16* heads = (__bf16*)(ws + 24 * MB);    //  8 MB [b,t,d]
    __bf16* Qb    = (__bf16*)(ws + 32 * MB);    //  8 MB bf16 copy of Q
    __bf16* Kb    = (__bf16*)(ws + 40 * MB);    //  8 MB bf16 copy of K
    __bf16* Vb    = (__bf16*)(ws + 48 * MB);    //  8 MB bf16 copy of V
    __bf16* WQb   = (__bf16*)(ws + 56 * MB);    //  2 MB
    __bf16* WKb   = (__bf16*)(ws + 58 * MB);    //  2 MB
    __bf16* WVb   = (__bf16*)(ws + 60 * MB);    //  2 MB
    __bf16* WOb   = (__bf16*)(ws + 62 * MB);    //  2 MB

    dim3 blk128(128), blk256(256);

    // 0) one-shot f32 -> bf16 conversions (inputs + weights)
    const int NX8 = BB * TT * DD / 8;   // 524288
    const int NW8 = DD * DD / 8;        // 131072
    cvt_kernel<<<dim3(NX8 / 256), blk256, 0, stream>>>(Q,  Qb,  NX8);
    cvt_kernel<<<dim3(NX8 / 256), blk256, 0, stream>>>(K,  Kb,  NX8);
    cvt_kernel<<<dim3(NX8 / 256), blk256, 0, stream>>>(V,  Vb,  NX8);
    cvt_kernel<<<dim3(NW8 / 256), blk256, 0, stream>>>(WQ, WQb, NW8);
    cvt_kernel<<<dim3(NW8 / 256), blk256, 0, stream>>>(WK, WKb, NW8);
    cvt_kernel<<<dim3(NW8 / 256), blk256, 0, stream>>>(WV, WVb, NW8);
    cvt_kernel<<<dim3(NW8 / 256), blk256, 0, stream>>>(WO, WOb, NW8);

    // 1) projections -> bf16 (q,k row-major per head; v transposed per head)
    dim3 gproj(DD / 64 / 4, (BB * TT) / 16, 1);
    proj_kernel<false><<<gproj, blk128, 0, stream>>>(Qb, WQb, qb);
    proj_kernel<false><<<gproj, blk128, 0, stream>>>(Kb, WKb, kb);
    proj_kernel<true ><<<gproj, blk128, 0, stream>>>(Vb, WVb, vt);

    // 2) masked scores -> attn (f32, raw, non-temporal)
    dim3 gscore(TT / 64 / 4, TT / 16, BB * HH);
    scores_kernel<<<gscore, blk128, 0, stream>>>(qb, kb, mask, attn);

    // 3) softmax in place
    dim3 gsm(TT, BB * HH);
    softmax_kernel<<<gsm, blk256, 0, stream>>>(attn);

    // 4) attn @ v -> heads bf16 [B,T,D]
    dim3 gav(1, TT / 16 / 4, BB * HH);
    av_kernel<<<gav, blk128, 0, stream>>>(attn, vt, heads);

    // 5) output projection (+ eval-mode dropout scale)
    dim3 gop(DD / 64 / 4, (BB * TT) / 16, 1);
    oproj_kernel<<<gop, blk128, 0, stream>>>(heads, WOb, out);
}
